// Attention_89120571392536
// MI455X (gfx1250) — matrix operand order repaired and checked by reading, not compiled, not run on hardware
//
#include <hip/hip_runtime.h>

// DIN attention pooling for MI455X (gfx1250, wave32).
// Memory-bound: hist = 210 MB @ 23.3 TB/s ~ 9us floor. GEMM folded to K=64
// effective-weight form, executed with v_wmma_f32_16x16x32_f16.
// Epilogue writes h to LDS directly (no cross-lane reductions / atomics).

typedef __attribute__((ext_vector_type(16))) _Float16 v16h;
typedef __attribute__((ext_vector_type(8)))  _Float16 v8h;
typedef __attribute__((ext_vector_type(8)))  float    v8f;

#define HIST     200
#define SPAD     208      // 13 tiles of 16 rows
#define DIM      64
#define HIDDEN   32
#define HSTR     72       // f16 row stride (halves) for Hh/WbT: DWORD stride 36 -> conflict-free
#define HSTRH    40       // f16 row stride (halves) for h tile: 80B rows, 16B aligned
#define NEG_INF_F (-4294967295.0f)   // -2^32 + 1

__global__ __launch_bounds__(256) void din_attn_kernel(
    const float* __restrict__ cand,   // (B, 64)
    const float* __restrict__ hist,   // (B, 200, 64)
    const int*   __restrict__ lens,   // (B,)
    const float* __restrict__ W1,     // (256, 32)
    const float* __restrict__ B1,     // (32,)
    const float* __restrict__ W2,     // (32,)
    const float* __restrict__ B2,     // (1,)
    float*       __restrict__ out)    // (B, 64)
{
    __shared__ _Float16 Hh[SPAD * HSTR];        // 29,952 B: f16 hist tile
    __shared__ _Float16 WbT[HIDDEN * HSTR];     //  4,608 B: effective weights, n-major
    __shared__ _Float16 s_h[SPAD * HSTRH];      // 16,640 B: relu'd hidden activations
    __shared__ float s_w[SPAD];                 // softmax weights
    __shared__ float s_bias[HIDDEN];
    __shared__ float s_w2[HIDDEN];
    __shared__ float s_part[256];
    __shared__ float s_red[8];
    __shared__ float s_gmax, s_gsum, s_b2;
    __shared__ int   s_len;

    const int tid = threadIdx.x;
    const int b   = blockIdx.x;
    const float* hb = hist + (size_t)b * (HIST * DIM);
    const float* cb = cand + (size_t)b * DIM;

    // ---- Phase 1: stream hist tile HBM -> LDS (f32 -> f16), float4 coalesced ----
    {
        const float4* h4 = (const float4*)hb;
        for (int idx = tid; idx < HIST * DIM / 4; idx += 256) {
            float4 v = h4[idx];
            int s = idx >> 4;              // 16 float4 per 64-wide row
            int c = (idx & 15) << 2;
            _Float16* p = &Hh[s * HSTR + c];
            p[0] = (_Float16)v.x; p[1] = (_Float16)v.y;
            p[2] = (_Float16)v.z; p[3] = (_Float16)v.w;
        }
        for (int idx = tid; idx < (SPAD - HIST) * DIM; idx += 256) {
            int s = HIST + (idx >> 6);
            int c = idx & 63;
            Hh[s * HSTR + c] = (_Float16)0.0f;   // zero pad rows 200..207
        }
    }

    // ---- Phase 2: effective weights  Wb[k][n] = W1B - W1D + cand[k]*W1C ----
    // Stored transposed (n-major) so WMMA B-fragments are contiguous K runs.
    for (int e = tid; e < HIDDEN * DIM; e += 256) {
        int n = e & 31;
        int k = e >> 5;
        float w = W1[(64 + k) * 32 + n] - W1[(192 + k) * 32 + n]
                + cb[k] * W1[(128 + k) * 32 + n];
        WbT[n * HSTR + k] = (_Float16)w;
    }
    // bias_eff[h] = B1[h] + sum_k cand[k]*(W1A[k][h] + W1D[k][h])   (f32)
    if (tid < HIDDEN) {
        float acc = B1[tid];
        for (int k = 0; k < DIM; ++k)
            acc += cb[k] * (W1[k * 32 + tid] + W1[(192 + k) * 32 + tid]);
        s_bias[tid] = acc;
        s_w2[tid]   = W2[tid];
    }
    if (tid == 0) { s_b2 = B2[0]; s_len = lens[b]; }
    __syncthreads();

    // ---- Phase 3: WMMA GEMM  (SPAD x 64) @ (64 x 32) -> relu(h) into LDS ----
    const int lane = tid & 31;
    const int wid  = tid >> 5;
    const int half = lane >> 4;   // 0: lanes 0-15, 1: lanes 16-31
    const int l16  = lane & 15;

    for (int t = wid; t < 26; t += 8) {             // 13 M-tiles x 2 N-tiles
        const int m0 = (t >> 1) << 4;
        const int n0 = (t & 1) << 4;
        const _Float16* arow = &Hh[(m0 + l16) * HSTR];
        const _Float16* bcol = &WbT[(n0 + l16) * HSTR];
        // Hoist all fragment loads (8x ds_load_b128) ahead of the WMMAs.
        v8h a0lo = *(const v8h*)(arow + 8 * half);
        v8h a0hi = *(const v8h*)(arow + 16 + 8 * half);
        v8h b0lo = *(const v8h*)(bcol + 8 * half);
        v8h b0hi = *(const v8h*)(bcol + 16 + 8 * half);
        v8h a1lo = *(const v8h*)(arow + 32 + 8 * half);
        v8h a1hi = *(const v8h*)(arow + 48 + 8 * half);
        v8h b1lo = *(const v8h*)(bcol + 32 + 8 * half);
        v8h b1hi = *(const v8h*)(bcol + 48 + 8 * half);
        v16h A0, B0, A1, B1f;
#pragma unroll
        for (int i = 0; i < 8; ++i) {
            A0[i] = a0lo[i];  A0[i + 8] = a0hi[i];
            B0[i] = b0lo[i];  B0[i + 8] = b0hi[i];
            A1[i] = a1lo[i];  A1[i + 8] = a1hi[i];
            B1f[i] = b1lo[i]; B1f[i + 8] = b1hi[i];
        }
        v8f c = {};
        c = __builtin_amdgcn_wmma_f32_16x16x32_f16(
                false, A0, false, B0, (short)0, c, false, false);
        c = __builtin_amdgcn_wmma_f32_16x16x32_f16(
                false, A1, false, B1f, (short)0, c, false, false);
        // Epilogue: D element (vgpr i, lane) -> M = m0+i+8*half, N = n0+l16.
        // Each (M,N) owned by exactly one lane: plain ds_store_b16, no reduction.
        const float bn = s_bias[n0 + l16];
#pragma unroll
        for (int i = 0; i < 8; ++i) {
            float hv = fmaxf(c[i] + bn, 0.0f);
            s_h[(m0 + i + 8 * half) * HSTRH + n0 + l16] = (_Float16)hv;
        }
    }
    __syncthreads();

    // ---- Phase 4: scores[s] = h[s,:] . W2 + b2, masked softmax over 200 ----
    float acc = 0.0f;
    if (tid < SPAD) {
        const v8h* hp = (const v8h*)&s_h[tid * HSTRH];   // 16B-aligned rows
#pragma unroll
        for (int cch = 0; cch < 4; ++cch) {
            v8h v = hp[cch];
#pragma unroll
            for (int j = 0; j < 8; ++j)
                acc += (float)v[j] * s_w2[cch * 8 + j];
        }
    }
    float val;
    if (tid < HIST) {
        float sc = acc + s_b2;
        val = (tid < s_len) ? sc * 0.125f : NEG_INF_F * 0.125f;
    } else {
        val = -__builtin_inff();
    }
    float m = val;
    m = fmaxf(m, __shfl_xor(m, 1));
    m = fmaxf(m, __shfl_xor(m, 2));
    m = fmaxf(m, __shfl_xor(m, 4));
    m = fmaxf(m, __shfl_xor(m, 8));
    m = fmaxf(m, __shfl_xor(m, 16));
    if (lane == 0) s_red[wid] = m;
    __syncthreads();
    if (tid == 0) {
        float g = s_red[0];
        for (int i = 1; i < 8; ++i) g = fmaxf(g, s_red[i]);
        s_gmax = g;
    }
    __syncthreads();
    float e = (tid < HIST) ? __expf(val - s_gmax) : 0.0f;
    float ssum = e;
    ssum += __shfl_xor(ssum, 1);
    ssum += __shfl_xor(ssum, 2);
    ssum += __shfl_xor(ssum, 4);
    ssum += __shfl_xor(ssum, 8);
    ssum += __shfl_xor(ssum, 16);
    if (lane == 0) s_red[wid] = ssum;
    __syncthreads();
    if (tid == 0) {
        float g = 0.0f;
        for (int i = 0; i < 8; ++i) g += s_red[i];
        s_gsum = g;
    }
    __syncthreads();
    if (tid < HIST) s_w[tid] = e / s_gsum;
    __syncthreads();

    // ---- Phase 5: out[d] = sum_s w[s] * hist[s][d]  (f32, L2-resident re-read) ----
    {
        const int d = tid & 63;
        const int g = tid >> 6;
        float wacc = 0.0f;
        for (int s = g; s < HIST; s += 4)
            wacc += s_w[s] * hb[s * DIM + d];
        s_part[tid] = wacc;
        __syncthreads();
        if (tid < DIM)
            out[(size_t)b * DIM + tid] =
                s_part[tid] + s_part[tid + 64] + s_part[tid + 128] + s_part[tid + 192];
    }
}

extern "C" void kernel_launch(void* const* d_in, const int* in_sizes, int n_in,
                              void* d_out, int out_size, void* d_ws, size_t ws_size,
                              hipStream_t stream) {
    const float* cand = (const float*)d_in[0];
    const float* hist = (const float*)d_in[1];
    const int*   lens = (const int*)d_in[2];
    const float* W1   = (const float*)d_in[3];
    const float* B1   = (const float*)d_in[4];
    const float* W2   = (const float*)d_in[5];
    const float* B2   = (const float*)d_in[6];
    float* out = (float*)d_out;
    const int B = in_sizes[0] / DIM;   // 4096
    din_attn_kernel<<<B, 256, 0, stream>>>(cand, hist, lens, W1, B1, W2, B2, out);
}
